// MultiHeadAttention_81896436400761
// MI455X (gfx1250) — matrix operand order repaired and checked by reading, not compiled, not run on hardware
//
#include <hip/hip_runtime.h>

// ---------------------------------------------------------------------------
// MI455X (gfx1250) wave32 WMMA implementation of quantum multi-head attention.
// All GEMMs use v_wmma_f32_16x16x32_bf16 (bf16 operands, fp32 accumulate).
// ---------------------------------------------------------------------------

typedef __bf16 bf16_t;
typedef __attribute__((ext_vector_type(16))) __bf16 v16bf;
typedef __attribute__((ext_vector_type(8)))  __bf16 v8bf;
typedef __attribute__((ext_vector_type(8)))  float  v8f;

union V16 { v16bf v; v8bf h[2]; };

__device__ __forceinline__ v8bf load8g(const bf16_t* p) { return *(const v8bf*)p; }

__device__ __forceinline__ v8f wmma_bf16(v16bf a, v16bf b, v8f c) {
  // D = A(16x32 bf16) x B(32x16 bf16) + C(16x16 f32)
  return __builtin_amdgcn_wmma_f32_16x16x32_bf16(false, a, false, b, (short)0, c,
                                                 false, false);
}

constexpr int E_DIM = 1024;
constexpr int SEQ   = 2048;
constexpr int NHEAD = 16;
constexpr int HD    = 64;
constexpr int TOK   = 4 * SEQ;   // 8192 flattened tokens

// ---------------------------------------------------------------------------
// fp32 -> bf16 conversion (4-wide)
// ---------------------------------------------------------------------------
__global__ void cvt_kernel(const float* __restrict__ in, bf16_t* __restrict__ out,
                           int n4) {
  int i = blockIdx.x * blockDim.x + threadIdx.x;
  int stride = gridDim.x * blockDim.x;
  for (; i < n4; i += stride) {
    float4 v = ((const float4*)in)[i];
    union { bf16_t b[4]; uint2 u; } pk;
    pk.b[0] = (bf16_t)v.x; pk.b[1] = (bf16_t)v.y;
    pk.b[2] = (bf16_t)v.z; pk.b[3] = (bf16_t)v.w;
    ((uint2*)out)[i] = pk.u;
  }
}

// ---------------------------------------------------------------------------
// GEMM: C[M=8192, N=1024] = A[M,K=1024] @ W[N,K]^T + bias[N]
// MODE 0: bf16 row-major out   MODE 1: bf16 transposed-per-head (for V)
// MODE 2: fp32 row-major out (final projection into d_out)
// Macro-tile 128x128, 8 waves (4M x 2N), wave tile 32x64.
// Operands are L2-resident (<=17MB each, L2=192MB) -> direct global fragment
// loads; no LDS staging needed for a compute-bound GEMM on this chip.
// ---------------------------------------------------------------------------
template <int MODE>
__global__ void __launch_bounds__(256)
gemm_kernel(const bf16_t* __restrict__ A, const bf16_t* __restrict__ W,
            const float* __restrict__ bias, void* __restrict__ Out) {
  const int lane = threadIdx.x & 31;
  const int wave = threadIdx.x >> 5;
  const int hf   = lane >> 4;      // half-wave select (A: K-slice, B: K-slice)
  const int r    = lane & 15;
  const int wm   = wave & 3;
  const int wn   = wave >> 2;
  const int m0   = blockIdx.y * 128 + wm * 32;
  const int n0   = blockIdx.x * 128 + wn * 64;

  v8f acc[2][4];
#pragma unroll
  for (int mt = 0; mt < 2; ++mt)
#pragma unroll
    for (int nt = 0; nt < 4; ++nt)
#pragma unroll
      for (int g = 0; g < 8; ++g) acc[mt][nt][g] = 0.f;

  for (int k0 = 0; k0 < E_DIM; k0 += 32) {
    V16 af[2];
#pragma unroll
    for (int mt = 0; mt < 2; ++mt) {
      const bf16_t* pa = A + (size_t)(m0 + mt * 16 + r) * E_DIM + k0 + hf * 8;
      af[mt].h[0] = load8g(pa);
      af[mt].h[1] = load8g(pa + 16);
    }
#pragma unroll
    for (int nt = 0; nt < 4; ++nt) {
      V16 bf;
      const bf16_t* pb = W + (size_t)(n0 + nt * 16 + r) * E_DIM + k0 + hf * 8;
      bf.h[0] = load8g(pb);
      bf.h[1] = load8g(pb + 16);
#pragma unroll
      for (int mt = 0; mt < 2; ++mt)
        acc[mt][nt] = wmma_bf16(af[mt].v, bf.v, acc[mt][nt]);
    }
  }

#pragma unroll
  for (int mt = 0; mt < 2; ++mt)
#pragma unroll
    for (int nt = 0; nt < 4; ++nt) {
      const int n = n0 + nt * 16 + r;
      const float bv = bias[n];
#pragma unroll
      for (int g = 0; g < 8; ++g) {
        const int m = m0 + mt * 16 + g + hf * 8;
        const float val = acc[mt][nt][g] + bv;
        if (MODE == 0) {
          ((bf16_t*)Out)[(size_t)m * E_DIM + n] = (bf16_t)val;
        } else if (MODE == 1) {
          // Vt layout: ((b*E + n) * SEQ + s), n = h*64 + d
          const int b = m >> 11, s = m & (SEQ - 1);
          ((bf16_t*)Out)[((size_t)b * E_DIM + n) * SEQ + s] = (bf16_t)val;
        } else {
          ((float*)Out)[(size_t)m * E_DIM + n] = val;
        }
      }
    }
}

// ---------------------------------------------------------------------------
// Quantum layer (in place on projection output).  One wave per (16-token tile,
// head): pre-GEMM (8 WMMA) -> 6 qubit rotations performed directly in the
// WMMA C-register layout (qubits 0-3: shfl_xor across lanes; 4-5: across
// accumulator tiles) -> C->A relayout through a wave-private LDS slab (same-
// wave LDS ops are in-order; no barrier required) -> post-GEMM (8 WMMA) ->
// residual add -> store.
// ---------------------------------------------------------------------------
__global__ void __launch_bounds__(256)
quantum_kernel(bf16_t* __restrict__ Xh, const bf16_t* __restrict__ preW,
               const float* __restrict__ preb, const bf16_t* __restrict__ postW,
               const float* __restrict__ postb, const float* __restrict__ theta,
               const float* __restrict__ phi) {
  __shared__ bf16_t lds[8][16 * 64];
  const int lane = threadIdx.x & 31;
  const int wave = threadIdx.x >> 5;
  const int hf = lane >> 4, r = lane & 15;
  const int task = blockIdx.x * 8 + wave;   // 512 tiles * 16 heads = 8192
  const int h = task & 15;
  const int m0 = (task >> 4) * 16;
  const int c0 = h * HD;
  bf16_t* slab = &lds[wave][0];

  float cs[6], sn[6], cpv[6];
#pragma unroll
  for (int i = 0; i < 6; ++i) {
    const float th = theta[i];
    cs[i] = __cosf(th); sn[i] = __sinf(th); cpv[i] = __cosf(phi[i]);
  }

  // xf A-fragments (residual source too)
  V16 xf[2];
#pragma unroll
  for (int kk = 0; kk < 2; ++kk) {
    const bf16_t* p = Xh + (size_t)(m0 + r) * E_DIM + c0 + kk * 32 + hf * 8;
    xf[kk].h[0] = load8g(p);
    xf[kk].h[1] = load8g(p + 16);
  }

  // pre-GEMM: res = xf @ preW^T + preb
  v8f acc[4];
#pragma unroll
  for (int nt = 0; nt < 4; ++nt) {
    const float bv = preb[nt * 16 + r];
#pragma unroll
    for (int g = 0; g < 8; ++g) acc[nt][g] = bv;
  }
#pragma unroll
  for (int kk = 0; kk < 2; ++kk)
#pragma unroll
    for (int nt = 0; nt < 4; ++nt) {
      V16 bw;
      const bf16_t* pb = preW + (nt * 16 + r) * HD + kk * 32 + hf * 8;
      bw.h[0] = load8g(pb);
      bw.h[1] = load8g(pb + 16);
      acc[nt] = wmma_bf16(xf[kk].v, bw.v, acc[nt]);
    }

  // qubits 0-3: column-pair rotation across lanes (column = nt*16 + r)
#pragma unroll
  for (int i = 0; i < 4; ++i) {
    const int mask = 1 << i;
    const float c = cs[i], s = sn[i], cp = cpv[i];
#pragma unroll
    for (int nt = 0; nt < 4; ++nt)
#pragma unroll
      for (int g = 0; g < 8; ++g) {
        const float v = acc[nt][g];
        const float o = __shfl_xor(v, mask, 32);
        acc[nt][g] = ((r & mask) == 0) ? (c * v - s * o) : ((s * o + c * v) * cp);
      }
  }
  // qubit 4: tile pairs (0,1),(2,3)
  {
    const float c = cs[4], s = sn[4], cp = cpv[4];
#pragma unroll
    for (int tp = 0; tp < 2; ++tp)
#pragma unroll
      for (int g = 0; g < 8; ++g) {
        const float x0 = acc[tp * 2][g], x1 = acc[tp * 2 + 1][g];
        acc[tp * 2][g] = c * x0 - s * x1;
        acc[tp * 2 + 1][g] = (s * x0 + c * x1) * cp;
      }
  }
  // qubit 5: tile pairs (0,2),(1,3)
  {
    const float c = cs[5], s = sn[5], cp = cpv[5];
#pragma unroll
    for (int tp = 0; tp < 2; ++tp)
#pragma unroll
      for (int g = 0; g < 8; ++g) {
        const float x0 = acc[tp][g], x1 = acc[tp + 2][g];
        acc[tp][g] = c * x0 - s * x1;
        acc[tp + 2][g] = (s * x0 + c * x1) * cp;
      }
  }

  // C-layout -> LDS (row-major 16x64) -> A-fragments (same-wave: in order)
#pragma unroll
  for (int nt = 0; nt < 4; ++nt)
#pragma unroll
    for (int g = 0; g < 8; ++g)
      slab[(g + hf * 8) * HD + nt * 16 + r] = (bf16_t)acc[nt][g];

  V16 rf[2];
#pragma unroll
  for (int kk = 0; kk < 2; ++kk) {
    const bf16_t* p = slab + r * HD + kk * 32 + hf * 8;
    rf[kk].h[0] = *(const v8bf*)p;
    rf[kk].h[1] = *(const v8bf*)(p + 16);
  }

  // post-GEMM: out = res @ postW^T + postb + xf
  v8f acc2[4];
#pragma unroll
  for (int nt = 0; nt < 4; ++nt) {
    const float bv = postb[nt * 16 + r];
#pragma unroll
    for (int g = 0; g < 8; ++g) acc2[nt][g] = bv;
  }
#pragma unroll
  for (int kk = 0; kk < 2; ++kk)
#pragma unroll
    for (int nt = 0; nt < 4; ++nt) {
      V16 bw;
      const bf16_t* pb = postW + (nt * 16 + r) * HD + kk * 32 + hf * 8;
      bw.h[0] = load8g(pb);
      bw.h[1] = load8g(pb + 16);
      acc2[nt] = wmma_bf16(rf[kk].v, bw.v, acc2[nt]);
    }

#pragma unroll
  for (int nt = 0; nt < 4; ++nt)
#pragma unroll
    for (int g = 0; g < 8; ++g) {
      const size_t idx = (size_t)(m0 + g + hf * 8) * E_DIM + c0 + nt * 16 + r;
      const float res = acc2[nt][g] + (float)Xh[idx];  // residual
      Xh[idx] = (bf16_t)res;                           // in-place (wave-private)
    }
}

// ---------------------------------------------------------------------------
// Flash attention: one wave per (b, h, 16-query tile).  Online softmax in
// fp32; scores and PV via WMMA.  V is pre-transposed per head so PV
// B-fragments are contiguous.  P relayout via wave-private LDS slab.
// ---------------------------------------------------------------------------
__global__ void __launch_bounds__(256)
flash_kernel(const bf16_t* __restrict__ Q, const bf16_t* __restrict__ K,
             const bf16_t* __restrict__ Vt, bf16_t* __restrict__ O) {
  __shared__ bf16_t plds[8][16 * 32];
  const int lane = threadIdx.x & 31;
  const int wave = threadIdx.x >> 5;
  const int hf = lane >> 4, r = lane & 15;
  const int task = blockIdx.x * 8 + wave;         // 4*16*128 = 8192 tasks
  const int qt = task & 127;
  const int h  = (task >> 7) & 15;
  const int b  = task >> 11;
  const int q0 = qt * 16;
  bf16_t* slab = &plds[wave][0];

  // Q A-fragments, pre-scaled by 1/sqrt(Dh) = 0.125
  V16 qf[2];
#pragma unroll
  for (int kk = 0; kk < 2; ++kk) {
    const bf16_t* p =
        Q + (size_t)(b * SEQ + q0 + r) * E_DIM + h * HD + kk * 32 + hf * 8;
    qf[kk].h[0] = load8g(p);
    qf[kk].h[1] = load8g(p + 16);
#pragma unroll
    for (int e = 0; e < 16; ++e)
      qf[kk].v[e] = (bf16_t)((float)qf[kk].v[e] * 0.125f);
  }

  v8f o[4];
  float mi[8], li[8];
#pragma unroll
  for (int nt = 0; nt < 4; ++nt)
#pragma unroll
    for (int g = 0; g < 8; ++g) o[nt][g] = 0.f;
#pragma unroll
  for (int g = 0; g < 8; ++g) { mi[g] = -1e30f; li[g] = 0.f; }

  const bf16_t* Kbase = K + (size_t)b * SEQ * E_DIM + h * HD;
  const bf16_t* Vbase = Vt + ((size_t)b * E_DIM + h * HD) * SEQ;

  for (int kv0 = 0; kv0 < SEQ; kv0 += 32) {
    // scores S (16x32) = Q (16x64) @ K^T
    v8f sacc[2];
#pragma unroll
    for (int st = 0; st < 2; ++st)
#pragma unroll
      for (int g = 0; g < 8; ++g) sacc[st][g] = 0.f;
#pragma unroll
    for (int kk = 0; kk < 2; ++kk)
#pragma unroll
      for (int st = 0; st < 2; ++st) {
        V16 kf;
        const bf16_t* pk =
            Kbase + (size_t)(kv0 + st * 16 + r) * E_DIM + kk * 32 + hf * 8;
        kf.h[0] = load8g(pk);
        kf.h[1] = load8g(pk + 16);
        sacc[st] = wmma_bf16(qf[kk].v, kf.v, sacc[st]);
      }

    // online softmax update (row = g + hf*8, 32 columns per row)
    float alpha[8];
#pragma unroll
    for (int g = 0; g < 8; ++g) {
      float mx = fmaxf(sacc[0][g], sacc[1][g]);
#pragma unroll
      for (int d = 1; d < 16; d <<= 1) mx = fmaxf(mx, __shfl_xor(mx, d, 32));
      const float mnew = fmaxf(mi[g], mx);
      alpha[g] = __expf(mi[g] - mnew);
      const float p0 = __expf(sacc[0][g] - mnew);
      const float p1 = __expf(sacc[1][g] - mnew);
      sacc[0][g] = p0;
      sacc[1][g] = p1;
      float rs = p0 + p1;
#pragma unroll
      for (int d = 1; d < 16; d <<= 1) rs += __shfl_xor(rs, d, 32);
      li[g] = li[g] * alpha[g] + rs;
      mi[g] = mnew;
    }
#pragma unroll
    for (int nt = 0; nt < 4; ++nt)
#pragma unroll
      for (int g = 0; g < 8; ++g) o[nt][g] *= alpha[g];

    // P: C-layout -> LDS -> A-fragment (wave-private slab, in-order LDS)
#pragma unroll
    for (int st = 0; st < 2; ++st)
#pragma unroll
      for (int g = 0; g < 8; ++g)
        slab[(g + hf * 8) * 32 + st * 16 + r] = (bf16_t)sacc[st][g];
    V16 pf;
    {
      const bf16_t* pp = slab + r * 32 + hf * 8;
      pf.h[0] = *(const v8bf*)pp;
      pf.h[1] = *(const v8bf*)(pp + 16);
    }

    // O += P (16x32) @ V (32x64); Vt rows (= head dims) are contiguous in kv
#pragma unroll
    for (int nt = 0; nt < 4; ++nt) {
      V16 vf;
      const bf16_t* pv = Vbase + (size_t)(nt * 16 + r) * SEQ + kv0 + hf * 8;
      vf.h[0] = load8g(pv);
      vf.h[1] = load8g(pv + 16);
      o[nt] = wmma_bf16(pf.v, vf.v, o[nt]);
    }
  }

  // normalize and store bf16 row-major (token, e)
#pragma unroll
  for (int g = 0; g < 8; ++g) {
    const float inv = 1.0f / li[g];
    const size_t base = (size_t)(b * SEQ + q0 + g + hf * 8) * E_DIM + h * HD;
#pragma unroll
    for (int nt = 0; nt < 4; ++nt)
      O[base + nt * 16 + r] = (bf16_t)(o[nt][g] * inv);
  }
}

// ---------------------------------------------------------------------------
// Host orchestration
// ---------------------------------------------------------------------------
extern "C" void kernel_launch(void* const* d_in, const int* in_sizes, int n_in,
                              void* d_out, int out_size, void* d_ws,
                              size_t ws_size, hipStream_t stream) {
  const float* query   = (const float*)d_in[0];
  const float* key     = (const float*)d_in[1];
  const float* value   = (const float*)d_in[2];
  const float* wq      = (const float*)d_in[3];
  const float* bq      = (const float*)d_in[4];
  const float* wk      = (const float*)d_in[5];
  const float* bk      = (const float*)d_in[6];
  const float* wv      = (const float*)d_in[7];
  const float* bv      = (const float*)d_in[8];
  const float* wo      = (const float*)d_in[9];
  const float* bo      = (const float*)d_in[10];
  const float* q_theta = (const float*)d_in[11];
  const float* q_phi   = (const float*)d_in[12];
  const float* q_preW  = (const float*)d_in[13];
  const float* q_preb  = (const float*)d_in[14];
  const float* q_postW = (const float*)d_in[15];
  const float* q_postb = (const float*)d_in[16];
  const float* k_theta = (const float*)d_in[17];
  const float* k_phi   = (const float*)d_in[18];
  const float* k_preW  = (const float*)d_in[19];
  const float* k_preb  = (const float*)d_in[20];
  const float* k_postW = (const float*)d_in[21];
  const float* k_postb = (const float*)d_in[22];

  const size_t TE = (size_t)TOK * E_DIM;   // 8388608
  const size_t EE = (size_t)E_DIM * E_DIM; // 1048576
  bf16_t* ws      = (bf16_t*)d_ws;
  bf16_t* X       = ws;            // staging for bf16 activations; later AO
  bf16_t* Qp      = ws + TE;
  bf16_t* Kp      = ws + 2 * TE;
  bf16_t* Vt      = ws + 3 * TE;
  bf16_t* Wq      = ws + 4 * TE;
  bf16_t* Wk      = Wq + EE;
  bf16_t* Wv      = Wk + EE;
  bf16_t* Wo      = Wv + EE;
  bf16_t* preWq   = Wo + EE;
  bf16_t* postWq  = preWq + HD * HD;
  bf16_t* preWk   = postWq + HD * HD;
  bf16_t* postWk  = preWk + HD * HD;

  const dim3 blk(256);
  auto cvt = [&](const float* in, bf16_t* out, size_t n) {
    const int n4 = (int)(n / 4);
    int grid = (n4 + 255) / 256;
    if (grid > 2048) grid = 2048;
    cvt_kernel<<<grid, blk, 0, stream>>>(in, out, n4);
  };

  // weights -> bf16 (one-time, tiny vs. compute)
  cvt(wq, Wq, EE);
  cvt(wk, Wk, EE);
  cvt(wv, Wv, EE);
  cvt(wo, Wo, EE);
  cvt(q_preW, preWq, HD * HD);
  cvt(q_postW, postWq, HD * HD);
  cvt(k_preW, preWk, HD * HD);
  cvt(k_postW, postWk, HD * HD);

  const dim3 ggrid(E_DIM / 128, TOK / 128);  // 8 x 64

  // Q projection + quantum layer
  cvt(query, X, TE);
  gemm_kernel<0><<<ggrid, blk, 0, stream>>>(X, Wq, bq, Qp);
  quantum_kernel<<<1024, blk, 0, stream>>>(Qp, preWq, q_preb, postWq, q_postb,
                                           q_theta, q_phi);
  // K projection + quantum layer
  cvt(key, X, TE);
  gemm_kernel<0><<<ggrid, blk, 0, stream>>>(X, Wk, bk, Kp);
  quantum_kernel<<<1024, blk, 0, stream>>>(Kp, preWk, k_preb, postWk, k_postb,
                                           k_theta, k_phi);
  // V projection (stored transposed per head for PV fragment loads)
  cvt(value, X, TE);
  gemm_kernel<1><<<ggrid, blk, 0, stream>>>(X, Wv, bv, Vt);

  // attention (X is dead now; reuse as attention output buffer)
  flash_kernel<<<1024, blk, 0, stream>>>(Qp, Kp, Vt, X);

  // final projection -> fp32 d_out
  gemm_kernel<2><<<ggrid, blk, 0, stream>>>(X, Wo, bo, (float*)d_out);
}
